// MEGAFeatureExtractor_32830730010629
// MI455X (gfx1250) — matrix operand order repaired and checked by reading, not compiled, not run on hardware
//
#include <hip/hip_runtime.h>
#include <stdint.h>

// ---------------------------------------------------------------------------
// MEGA relation-attention for MI455X (gfx1250, wave32, WMMA bf16 16x16x32).
// Reassociated attention: V = ref_feat @ Wv_flat^T first, then attn @ V.
// Uses CDNA5 async global->LDS copies (ASYNCcnt) for GEMM A-tile staging.
// ---------------------------------------------------------------------------

typedef __bf16 bf16;
typedef __attribute__((ext_vector_type(16))) __bf16 v16bf;
typedef __attribute__((ext_vector_type(8)))  __bf16 v8bf;
typedef __attribute__((ext_vector_type(8)))  float  v8f;

#define NN   512
#define MMM  2048
#define FEAT 1024
#define GRP  16
#define DGD  64
#define LDA  1032   // padded LDS leading dim for A staging (8-elem (16 B) pad / 4-bank skew)

__device__ __forceinline__ v8f zero8() {
  v8f z;
#pragma unroll
  for (int i = 0; i < 8; ++i) z[i] = 0.0f;
  return z;
}

__device__ __forceinline__ v8f wmma_bf16(v16bf a, v16bf b, v8f c) {
  // D = A(16x32 bf16) * B(32x16 bf16) + C(16x16 f32)
  return __builtin_amdgcn_wmma_f32_16x16x32_bf16(false, a, false, b,
                                                 (short)0, c, false, false);
}

// CDNA5 async copy: 16B per lane, global -> LDS, tracked by ASYNCcnt.
__device__ __forceinline__ void async_ld_b128(uint32_t lds_byte_off,
                                              const void* saddr_base,
                                              uint32_t voffset_bytes) {
  asm volatile("global_load_async_to_lds_b128 %0, %1, %2"
               :: "v"(lds_byte_off), "v"(voffset_bytes), "s"(saddr_base)
               : "memory");
}
__device__ __forceinline__ void wait_async0() {
  asm volatile("s_wait_asynccnt 0x0" ::: "memory");
}

// A fragment: 16x32 bf16 tile, row-major source `base` with leading dim ld.
// (contiguous K order: lanes<16 khi=0, lanes>=16 khi=16)
__device__ __forceinline__ v16bf load_a(const bf16* base, int ld) {
  int lane = threadIdx.x & 31;
  int row  = lane & 15;
  int khi  = (lane >> 4) << 4;               // 0 or 16
  const bf16* p = base + row * ld + khi;
  v8bf lo = *(const v8bf*)(p);               // K = khi..khi+7
  v8bf hi = *(const v8bf*)(p + 8);           // K = khi+8..khi+15
  v16bf a;
#pragma unroll
  for (int i = 0; i < 8; ++i) { a[i] = lo[i]; a[i + 8] = hi[i]; }
  return a;
}

// B fragment where B[k][n] = base[n*ld + k]  (X @ W^T with W row-major).
// Lane n<16: K=0..15, lanes 16..31: K=16..31 -> contiguous 32B per lane.
__device__ __forceinline__ v16bf load_b_t(const bf16* base, int ld) {
  int lane = threadIdx.x & 31;
  int col  = lane & 15;
  int koff = (lane >> 4) << 4;               // 0 or 16
  return *(const v16bf*)(base + col * ld + koff);
}

// ---------------------------------------------------------------------------
__global__ __launch_bounds__(256) void cvt_kernel(const float* __restrict__ s,
                                                  bf16* __restrict__ d, int n) {
  int i = blockIdx.x * 256 + threadIdx.x;
  if (i < n) d[i] = (bf16)s[i];
}

// ---------------------------------------------------------------------------
// C[r,c] = sum_k A[r,k] * W[c,k] (+bias[c]); A rows x 1024, W 1024x1024.
// A 16-row block async-staged to LDS once, reused by all 8 waves (8 c-tiles).
__global__ __launch_bounds__(256) void gemm_nt(const bf16* __restrict__ A,
                                               const bf16* __restrict__ W,
                                               const float* __restrict__ bias,
                                               bf16* __restrict__ C) {
  __shared__ bf16 sA[16 * LDA];              // 32.25 KB
  int tid  = threadIdx.x;
  int wave = tid >> 5;
  int lane = tid & 31;
  int r0 = blockIdx.x * 16;
  int c0 = (blockIdx.y * 8 + wave) * 16;

  // async stage A[r0..r0+15][0..1023] -> sA (padded rows)
  {
    const bf16* abase = A + (size_t)r0 * FEAT;
    uint32_t sA0  = (uint32_t)(uintptr_t)&sA[0];
    int arow = tid >> 4, aseg = tid & 15;
#pragma unroll
    for (int i = 0; i < 8; ++i) {
      uint32_t goff = (uint32_t)((arow * FEAT + aseg * 64 + i * 8) * 2);
      uint32_t loff = sA0 + (uint32_t)((arow * LDA + aseg * 64 + i * 8) * 2);
      async_ld_b128(loff, abase, goff);
    }
    wait_async0();
    __syncthreads();
  }

  v8f acc = zero8();
  for (int k0 = 0; k0 < FEAT; k0 += 32) {
    v16bf a = load_a(sA + k0, LDA);                      // from LDS
    v16bf b = load_b_t(W + (size_t)c0 * FEAT + k0, FEAT);
    acc = wmma_bf16(a, b, acc);
  }
  int col   = lane & 15;
  int rbase = (lane >> 4) << 3;
  float bv = bias ? bias[c0 + col] : 0.0f;
#pragma unroll
  for (int r = 0; r < 8; ++r)
    C[(size_t)(r0 + rbase + r) * FEAT + c0 + col] = (bf16)(acc[r] + bv);
}

// ---------------------------------------------------------------------------
// uk[g,m] = u[g,:] . k_proj[m, g*64:(g+1)*64]
__global__ __launch_bounds__(256) void uk_kernel(const float* __restrict__ u,
                                                 const bf16* __restrict__ K,
                                                 float* __restrict__ UK) {
  int g = blockIdx.y;
  int m = blockIdx.x * 256 + threadIdx.x;
  float acc = 0.0f;
#pragma unroll 8
  for (int d = 0; d < DGD; ++d)
    acc += u[g * DGD + d] * (float)K[(size_t)m * FEAT + g * DGD + d];
  UK[g * MMM + m] = acc;
}

// ---------------------------------------------------------------------------
// posb[n,g,m] = log(relu(Wg . emb(n,m) + b_g) + 1e-6)   (bf16)
__global__ __launch_bounds__(128) void posb_kernel(const float* __restrict__ bbox,
                                                   const float* __restrict__ rbox,
                                                   const float* __restrict__ Wg_w,
                                                   const float* __restrict__ Wg_b,
                                                   bf16* __restrict__ PB) {
  __shared__ float sW[GRP][64];
  __shared__ float sB[GRP];
  int tid = threadIdx.x;
  for (int i = tid; i < GRP * 64; i += 128) sW[i >> 6][i & 63] = Wg_w[i];
  if (tid < GRP) sB[tid] = Wg_b[tid];

  int n = blockIdx.y;
  int m = blockIdx.x * 128 + tid;

  float x0 = bbox[n * 4 + 0], y0 = bbox[n * 4 + 1];
  float x1 = bbox[n * 4 + 2], y1 = bbox[n * 4 + 3];
  float w = x1 - x0 + 1.0f, h = y1 - y0 + 1.0f;
  float cx = 0.5f * (x0 + x1), cy = 0.5f * (y0 + y1);
  float rx0 = rbox[m * 4 + 0], ry0 = rbox[m * 4 + 1];
  float rx1 = rbox[m * 4 + 2], ry1 = rbox[m * 4 + 3];
  float wr = rx1 - rx0 + 1.0f, hr = ry1 - ry0 + 1.0f;
  float cxr = 0.5f * (rx0 + rx1), cyr = 0.5f * (ry0 + ry1);

  float dlt[4];
  dlt[0] = logf(fabsf((cx - cxr) / w) + 1e-3f);
  dlt[1] = logf(fabsf((cy - cyr) / h) + 1e-3f);
  dlt[2] = logf(w / wr);
  dlt[3] = logf(h / hr);

  // 100 / 1000^(f/8), f = 0..7
  const float invdim[8] = {100.0f, 42.1696503f, 17.7827941f, 7.49894209f,
                           3.16227766f, 1.33352143f, 0.562341325f, 0.237137371f};

  __syncthreads();

  float emb[64];
#pragma unroll
  for (int d = 0; d < 4; ++d) {
#pragma unroll
    for (int f = 0; f < 8; ++f) {
      float v = dlt[d] * invdim[f];
      emb[d * 16 + f]     = __sinf(v);
      emb[d * 16 + 8 + f] = __cosf(v);
    }
  }
#pragma unroll 1
  for (int g = 0; g < GRP; ++g) {
    float acc = sB[g];
#pragma unroll
    for (int e = 0; e < 64; ++e) acc += sW[g][e] * emb[e];
    PB[((size_t)n * GRP + g) * MMM + m] = (bf16)logf(fmaxf(acc, 0.0f) + 1e-6f);
  }
}

// ---------------------------------------------------------------------------
// Per (head g, 16-row tile): S = (QK^T + u.K^T)/8 + posb (WMMA), softmax over
// the full 2048-wide row held in LDS, then O = P @ V (WMMA, K split over
// wave pairs, V staged transposed in LDS) + Wv_b.
__global__ __launch_bounds__(256) void attn_kernel(const bf16* __restrict__ Q,
                                                   const bf16* __restrict__ K,
                                                   const bf16* __restrict__ V,
                                                   const float* __restrict__ UK,
                                                   const bf16* __restrict__ PB,
                                                   const float* __restrict__ Vb,
                                                   float* __restrict__ O) {
  __shared__ bf16  sP[16][MMM];     // 64 KB: scores, then probabilities
  __shared__ bf16  sVt[2][64][40];  // V chunk, transposed (d-major), padded
  __shared__ float sO[4][16][16];   // partial O from K-half 1
  __shared__ float sRed[16][16];
  __shared__ float sMax[16];
  __shared__ float sSum[16];

  int g  = blockIdx.x;
  int n0 = blockIdx.y * 16;
  int tid = threadIdx.x, wave = tid >> 5, lane = tid & 31;
  int col = lane & 15, rbase = (lane >> 4) << 3;

  // ---- phase 1: scores ---------------------------------------------------
  v16bf a0 = load_a(Q + (size_t)n0 * FEAT + g * DGD, FEAT);
  v16bf a1 = load_a(Q + (size_t)n0 * FEAT + g * DGD + 32, FEAT);
  for (int mt = wave; mt < MMM / 16; mt += 8) {
    int m0 = mt * 16;
    v16bf b0 = load_b_t(K + (size_t)m0 * FEAT + g * DGD, FEAT);
    v16bf b1 = load_b_t(K + (size_t)m0 * FEAT + g * DGD + 32, FEAT);
    v8f c = zero8();
    c = wmma_bf16(a0, b0, c);
    c = wmma_bf16(a1, b1, c);
#pragma unroll
    for (int r = 0; r < 8; ++r) {
      int nl = rbase + r, m = m0 + col;
      float s = (c[r] + UK[g * MMM + m]) * 0.125f
              + (float)PB[((size_t)(n0 + nl) * GRP + g) * MMM + m];
      sP[nl][m] = (bf16)s;
    }
  }
  __syncthreads();

  // ---- phase 2: softmax over m (row held entirely in LDS) ----------------
  int row = tid >> 4, cc = tid & 15;
  float mx = -3.0e38f;
  for (int m = cc; m < MMM; m += 16) mx = fmaxf(mx, (float)sP[row][m]);
  sRed[row][cc] = mx;
  __syncthreads();
  if (cc == 0) {
    float v = -3.0e38f;
#pragma unroll
    for (int i = 0; i < 16; ++i) v = fmaxf(v, sRed[row][i]);
    sMax[row] = v;
  }
  __syncthreads();
  float rmax = sMax[row], sum = 0.0f;
  for (int m = cc; m < MMM; m += 16) {
    float e = __expf((float)sP[row][m] - rmax);
    sum += e;
    sP[row][m] = (bf16)e;
  }
  sRed[row][cc] = sum;
  __syncthreads();
  if (cc == 0) {
    float v = 0.0f;
#pragma unroll
    for (int i = 0; i < 16; ++i) v += sRed[row][i];
    sSum[row] = v;
  }
  __syncthreads();
  float inv = 1.0f / sSum[row];
  for (int m = cc; m < MMM; m += 16)
    sP[row][m] = (bf16)((float)sP[row][m] * inv);
  __syncthreads();

  // ---- phase 3: O = P @ V ------------------------------------------------
  // Waves 0-3: K in [0,1024), d-tile = wave*16.  Waves 4-7: K in [1024,2048),
  // same d-tiles; partials combined through LDS.
  {
    int half = wave >> 2;
    int d0   = (wave & 3) * 16;
    int kk   = tid >> 3;            // 0..31  (staging row within chunk)
    int dsv  = (tid & 7) * 8;       // 0..56  (staging d start)
    int koff = (lane >> 4) << 4;    // 0 or 16
    v8f o = zero8();
    for (int it = 0; it < 32; ++it) {
      int kc0 = it * 32;
      // cooperative staging of both K-half chunks, transposed into LDS
      v8bf v0 = *(const v8bf*)(V + (size_t)(kc0 + kk) * FEAT + g * DGD + dsv);
      v8bf v1 = *(const v8bf*)(V + (size_t)(1024 + kc0 + kk) * FEAT + g * DGD + dsv);
#pragma unroll
      for (int i = 0; i < 8; ++i) {
        sVt[0][dsv + i][kk] = v0[i];
        sVt[1][dsv + i][kk] = v1[i];
      }
      __syncthreads();
      v16bf a = load_a(&sP[0][0] + half * 1024 + kc0, MMM);  // P from LDS
      const bf16* bp = &sVt[half][d0 + col][koff];
      v8bf blo = *(const v8bf*)(bp);
      v8bf bhi = *(const v8bf*)(bp + 8);
      v16bf b;
#pragma unroll
      for (int i = 0; i < 8; ++i) { b[i] = blo[i]; b[i + 8] = bhi[i]; }
      o = wmma_bf16(a, b, o);
      __syncthreads();
    }
    if (half == 1) {
#pragma unroll
      for (int r = 0; r < 8; ++r) sO[wave & 3][rbase + r][col] = o[r];
    }
    __syncthreads();
    if (half == 0) {
#pragma unroll
      for (int r = 0; r < 8; ++r) {
        int nl = rbase + r;
        O[(size_t)(n0 + nl) * FEAT + g * DGD + d0 + col] =
            o[r] + sO[wave][nl][col] + Vb[g * DGD + d0 + col];
      }
    }
  }
}

// ---------------------------------------------------------------------------
extern "C" void kernel_launch(void* const* d_in, const int* in_sizes, int n_in,
                              void* d_out, int out_size, void* d_ws, size_t ws_size,
                              hipStream_t stream) {
  (void)in_sizes; (void)n_in; (void)out_size; (void)ws_size;
  const float* bbox  = (const float*)d_in[0];
  const float* rbox  = (const float*)d_in[1];
  const float* roi   = (const float*)d_in[2];
  const float* ref   = (const float*)d_in[3];
  const float* Wg_w  = (const float*)d_in[4];
  const float* Wg_b  = (const float*)d_in[5];
  const float* Wq_w  = (const float*)d_in[6];
  const float* Wq_b  = (const float*)d_in[7];
  const float* Wk_w  = (const float*)d_in[8];
  const float* Wk_b  = (const float*)d_in[9];
  const float* Wv_w  = (const float*)d_in[10];   // [G,DG,FEAT] == [1024,1024]
  const float* Wv_b  = (const float*)d_in[11];
  const float* u     = (const float*)d_in[12];
  float* out = (float*)d_out;

  char* ws = (char*)d_ws;
  size_t off = 0;
  auto alloc = [&](size_t bytes) -> char* {
    char* p = ws + off;
    off += (bytes + 255) & ~(size_t)255;
    return p;
  };
  bf16*  roi_bf = (bf16*) alloc((size_t)NN  * FEAT * 2);
  bf16*  ref_bf = (bf16*) alloc((size_t)MMM * FEAT * 2);
  bf16*  wq_bf  = (bf16*) alloc((size_t)FEAT * FEAT * 2);
  bf16*  wk_bf  = (bf16*) alloc((size_t)FEAT * FEAT * 2);
  bf16*  wv_bf  = (bf16*) alloc((size_t)FEAT * FEAT * 2);
  bf16*  q_bf   = (bf16*) alloc((size_t)NN  * FEAT * 2);
  bf16*  k_bf   = (bf16*) alloc((size_t)MMM * FEAT * 2);
  bf16*  v_bf   = (bf16*) alloc((size_t)MMM * FEAT * 2);
  float* uk     = (float*)alloc((size_t)GRP * MMM * 4);
  bf16*  posb   = (bf16*) alloc((size_t)NN * GRP * MMM * 2);

  // fp32 -> bf16
  cvt_kernel<<<(NN  * FEAT + 255) / 256, 256, 0, stream>>>(roi,  roi_bf, NN  * FEAT);
  cvt_kernel<<<(MMM * FEAT + 255) / 256, 256, 0, stream>>>(ref,  ref_bf, MMM * FEAT);
  cvt_kernel<<<(FEAT * FEAT + 255) / 256, 256, 0, stream>>>(Wq_w, wq_bf, FEAT * FEAT);
  cvt_kernel<<<(FEAT * FEAT + 255) / 256, 256, 0, stream>>>(Wk_w, wk_bf, FEAT * FEAT);
  cvt_kernel<<<(FEAT * FEAT + 255) / 256, 256, 0, stream>>>(Wv_w, wv_bf, FEAT * FEAT);

  // projections (WMMA GEMMs, async-staged A tiles)
  gemm_nt<<<dim3(NN / 16, 8), 256, 0, stream>>>(roi_bf, wq_bf, Wq_b, q_bf);
  gemm_nt<<<dim3(MMM / 16, 8), 256, 0, stream>>>(ref_bf, wk_bf, Wk_b, k_bf);
  gemm_nt<<<dim3(MMM / 16, 8), 256, 0, stream>>>(ref_bf, wv_bf, (const float*)nullptr, v_bf);

  // content-bias term and geometric position bias
  uk_kernel<<<dim3(MMM / 256, GRP), 256, 0, stream>>>(u, k_bf, uk);
  posb_kernel<<<dim3(MMM / 128, NN), 128, 0, stream>>>(bbox, rbox, Wg_w, Wg_b, posb);

  // fused attention (scores + softmax + P@V)
  attn_kernel<<<dim3(GRP, NN / 16), 256, 0, stream>>>(q_bf, k_bf, v_bf, uk, posb,
                                                      Wv_b, out);
}
